// Swin_Block_38379827757321
// MI455X (gfx1250) — matrix operand order repaired and checked by reading, not compiled, not run on hardware
//
#include <hip/hip_runtime.h>
#include <hip/hip_bf16.h>
#include <math.h>

// ---------------------------------------------------------------------------
// Swin block, CDNA5 (gfx1250) WMMA implementation.
// B=16, H=W=56, C=384, WS=7, SHIFT=3, HEADS=12, HEAD_DIM=32
// Windows: 16*64 = 1024, tokens padded 49 -> 64.
// GEMMs: per-wave 2x4 register tile (M=32,N=64), double-buffered fragments.
// ---------------------------------------------------------------------------

typedef __attribute__((ext_vector_type(16))) _Float16 v16h;
typedef __attribute__((ext_vector_type(8)))  _Float16 v8h;
typedef __attribute__((ext_vector_type(8)))  float    v8f;

#define NWINB   1024        // 16 images * 64 windows
#define CDIM    384
#define NTOKP   64          // padded tokens per window
#define NTOK    49
#define HEADS   12
#define HDIM    32
#define NTOKENS 50176       // 16*3136
#define SCALE_Q 0.17677669529663687f

// ---- workspace layout (bytes) ---------------------------------------------
static constexpr size_t OFF_QKVT = 0;                       // 1152*384*2
static constexpr size_t OFF_PROJT = 884736;                 // 384*384*2
static constexpr size_t OFF_FC1T = 1179648;                 // 1536*384*2
static constexpr size_t OFF_FC2T = 2359296;                 // 384*1536*2
static constexpr size_t OFF_Y    = 3538944;                 // 1024*64*384*2 (aliased: attn O)
static constexpr size_t OFF_Q    = 53870592;                // 1024*12*64*32*2 (aliased: LN2 out)
static constexpr size_t OFF_K    = 70647808;
static constexpr size_t OFF_VT   = 87425024;
static constexpr size_t OFF_X1   = 104202240;               // 50176*384*4
static constexpr size_t OFF_FC1O = 181272576;               // 50176*1536*2

// ---------------------------------------------------------------------------
// WMMA fragment loads.
// A fragment (16x32 f16) from row-major [M][K]: lane m=L&15, g=L>>4,
//   halves v=0..7 -> K = 8g+v, v=8..15 -> K = 16+8g+(v-8).   (2x b128)
// B fragment (32x16 f16) from N-major [N][K]: lane n=L&15, g=L>>4,
//   halves v=0..7 -> K = 8g+v, v=8..15 -> K = 16+8g+(v-8).   (2x b128)
// ---------------------------------------------------------------------------
__device__ __forceinline__ v16h load_frag_a(const _Float16* __restrict__ base,
                                            int ld, int lane) {
  int m = lane & 15, g = lane >> 4;
  const _Float16* p = base + (size_t)m * ld + 8 * g;
  v8h lo = *(const v8h*)p;
  v8h hi = *(const v8h*)(p + 16);
  v16h r;
#pragma unroll
  for (int i = 0; i < 8; ++i) { r[i] = lo[i]; r[i + 8] = hi[i]; }
  return r;
}

__device__ __forceinline__ v16h load_frag_b(const _Float16* __restrict__ base,
                                            int ld, int lane) {
  int n = lane & 15, g = lane >> 4;
  const _Float16* p = base + (size_t)n * ld + 8 * g;
  v8h lo = *(const v8h*)p;
  v8h hi = *(const v8h*)(p + 16);
  v16h r;
#pragma unroll
  for (int i = 0; i < 8; ++i) { r[i] = lo[i]; r[i + 8] = hi[i]; }
  return r;
}

__device__ __forceinline__ v8f wmma16(v16h a, v16h b, v8f c) {
  return __builtin_amdgcn_wmma_f32_16x16x32_f16(false, a, false, b,
                                                (short)0, c, false, false);
}

__device__ __forceinline__ float wave_sum(float v) {
#pragma unroll
  for (int m = 16; m >= 1; m >>= 1) v += __shfl_xor(v, m, 32);
  return v;
}

__device__ __forceinline__ float gelu(float x) {
  return 0.5f * x * (1.0f + erff(x * 0.70710678118654752f));
}

// ---------------------------------------------------------------------------
// GEMM core: per-wave 2x4 tile (rows wave*32..+32, cols n0..n0+64),
// double-buffered fragment loads so global_load_b128 overlaps the WMMA chain.
// A: block base, row-major stride ld. B: Wt + n0*ld (N-major, stride ld).
// ---------------------------------------------------------------------------
__device__ __forceinline__ void gemm_core_2x4(const _Float16* __restrict__ A,
                                              const _Float16* __restrict__ B,
                                              int ld, int K, int wave, int lane,
                                              v8f acc[8]) {
  const _Float16* A0 = A + (size_t)(wave * 32) * ld;
  const _Float16* A1 = A0 + (size_t)16 * ld;
  v16h a0 = load_frag_a(A0, ld, lane);
  v16h a1 = load_frag_a(A1, ld, lane);
  v16h b0 = load_frag_b(B, ld, lane);
  v16h b1 = load_frag_b(B + (size_t)16 * ld, ld, lane);
  v16h b2 = load_frag_b(B + (size_t)32 * ld, ld, lane);
  v16h b3 = load_frag_b(B + (size_t)48 * ld, ld, lane);
  for (int k0 = 32; k0 < K; k0 += 32) {
    v16h na0 = load_frag_a(A0 + k0, ld, lane);
    v16h na1 = load_frag_a(A1 + k0, ld, lane);
    v16h nb0 = load_frag_b(B + k0, ld, lane);
    v16h nb1 = load_frag_b(B + (size_t)16 * ld + k0, ld, lane);
    v16h nb2 = load_frag_b(B + (size_t)32 * ld + k0, ld, lane);
    v16h nb3 = load_frag_b(B + (size_t)48 * ld + k0, ld, lane);
    acc[0] = wmma16(a0, b0, acc[0]);
    acc[1] = wmma16(a0, b1, acc[1]);
    acc[2] = wmma16(a0, b2, acc[2]);
    acc[3] = wmma16(a0, b3, acc[3]);
    acc[4] = wmma16(a1, b0, acc[4]);
    acc[5] = wmma16(a1, b1, acc[5]);
    acc[6] = wmma16(a1, b2, acc[6]);
    acc[7] = wmma16(a1, b3, acc[7]);
    a0 = na0; a1 = na1; b0 = nb0; b1 = nb1; b2 = nb2; b3 = nb3;
  }
  acc[0] = wmma16(a0, b0, acc[0]);
  acc[1] = wmma16(a0, b1, acc[1]);
  acc[2] = wmma16(a0, b2, acc[2]);
  acc[3] = wmma16(a0, b3, acc[3]);
  acc[4] = wmma16(a1, b0, acc[4]);
  acc[5] = wmma16(a1, b1, acc[5]);
  acc[6] = wmma16(a1, b2, acc[6]);
  acc[7] = wmma16(a1, b3, acc[7]);
}

// ---------------------------------------------------------------------------
// 0) weight transpose + fp32->f16:  Wt[n*K+k] = W[k*N+n]
// ---------------------------------------------------------------------------
__global__ __launch_bounds__(256) void wt_cvt_kernel(const float* __restrict__ W,
                                                     _Float16* __restrict__ Wt,
                                                     int K, int N) {
  int idx = blockIdx.x * 256 + threadIdx.x;
  if (idx >= K * N) return;
  int n = idx / K, k = idx - n * K;
  Wt[idx] = (_Float16)W[(size_t)k * N + n];
}

// ---------------------------------------------------------------------------
// 1) LN1 + roll(-3,-3) + window partition -> y f16 [1024][64][384], pad rows 0
// ---------------------------------------------------------------------------
__global__ __launch_bounds__(256) void ln1_window_kernel(
    const float* __restrict__ x, const float* __restrict__ gw,
    const float* __restrict__ bw, _Float16* __restrict__ y) {
  int win = blockIdx.x;
  int wave = threadIdx.x >> 5, lane = threadIdx.x & 31;
  int img = win >> 6, w_in = win & 63;
  int wi = w_in >> 3, wj = w_in & 7;
  for (int t = wave; t < NTOKP; t += 8) {
    _Float16* yrow = y + ((size_t)win * NTOKP + t) * CDIM;
    if (t >= NTOK) {
#pragma unroll
      for (int i = 0; i < 12; ++i) yrow[lane + 32 * i] = (_Float16)0.0f;
      continue;
    }
    int ti = t / 7, tj = t - ti * 7;
    int h = (wi * 7 + ti + 3) % 56;   // cyclic shift -3
    int w = (wj * 7 + tj + 3) % 56;
    const float* xrow = x + ((size_t)img * 3136 + h * 56 + w) * CDIM;
    float vals[12], s = 0.f, s2 = 0.f;
#pragma unroll
    for (int i = 0; i < 12; ++i) {
      float v = xrow[lane + 32 * i];
      vals[i] = v; s += v; s2 += v * v;
    }
    s = wave_sum(s); s2 = wave_sum(s2);
    float mu = s * (1.0f / CDIM);
    float rs = rsqrtf(s2 * (1.0f / CDIM) - mu * mu + 1e-5f);
#pragma unroll
    for (int i = 0; i < 12; ++i) {
      int c = lane + 32 * i;
      yrow[c] = (_Float16)((vals[i] - mu) * rs * gw[c] + bw[c]);
    }
  }
}

// ---------------------------------------------------------------------------
// 2) QKV GEMM: M=65536 (128/block) x N tile 64, K=384
//    -> q,k (token-major, q pre-scaled), v transposed
// ---------------------------------------------------------------------------
__global__ __launch_bounds__(128) void gemm_qkv_kernel(
    const _Float16* __restrict__ y, const _Float16* __restrict__ Wt,
    _Float16* __restrict__ qw, _Float16* __restrict__ kw,
    _Float16* __restrict__ vtw) {
  int n0 = blockIdx.y * 64;
  int wave = threadIdx.x >> 5, lane = threadIdx.x & 31;
  const _Float16* A = y + (size_t)blockIdx.x * 128 * CDIM;
  v8f acc[8] = {};
  gemm_core_2x4(A, Wt + (size_t)n0 * CDIM, CDIM, CDIM, wave, lane, acc);
  int g = lane >> 4, nl = lane & 15;
#pragma unroll
  for (int p = 0; p < 8; ++p) {
    int col = n0 + 16 * (p & 3) + nl;     // 0..1151
    int which = col / CDIM, rem = col - which * CDIM;
    int head = rem >> 5, hd = rem & 31;
#pragma unroll
    for (int r = 0; r < 8; ++r) {
      int mg = blockIdx.x * 128 + wave * 32 + (p >> 2) * 16 + g * 8 + r;
      int win = mg >> 6, tok = mg & 63;
      size_t hb = (size_t)win * HEADS + head;
      float v = acc[p][r];
      if (which == 0)      qw[(hb * NTOKP + tok) * HDIM + hd] = (_Float16)(v * SCALE_Q);
      else if (which == 1) kw[(hb * NTOKP + tok) * HDIM + hd] = (_Float16)v;
      else                 vtw[(hb * HDIM + hd) * NTOKP + tok] = (_Float16)v;
    }
  }
}

// ---------------------------------------------------------------------------
// 3) attention: one block per (window, head). S=q k^T (WMMA K=32),
//    mask+softmax in LDS, O = P V (WMMA), O -> [win][tok][head*32+hd]
// ---------------------------------------------------------------------------
__global__ __launch_bounds__(128) void attn_kernel(
    const _Float16* __restrict__ qw, const _Float16* __restrict__ kw,
    const _Float16* __restrict__ vtw, _Float16* __restrict__ o) {
  __shared__ __align__(16) _Float16 qs[NTOKP * HDIM];
  __shared__ __align__(16) _Float16 ks[NTOKP * HDIM];
  __shared__ __align__(16) _Float16 vts[HDIM * NTOKP];
  __shared__ __align__(16) float    S[NTOKP * NTOKP];
  __shared__ __align__(16) _Float16 P[NTOKP * NTOKP];
  __shared__ int lab[NTOK];

  int bid = blockIdx.x;
  int win = bid / HEADS, head = bid - win * HEADS;
  int w_in = win & 63;
  int wi = w_in >> 3, wj = w_in & 7;
  int tid = threadIdx.x;
  size_t base = ((size_t)win * HEADS + head) * (NTOKP * HDIM);

  {
    const uint4* qg = (const uint4*)(qw + base);
    const uint4* kg = (const uint4*)(kw + base);
    const uint4* vg = (const uint4*)(vtw + base);
    uint4* ql = (uint4*)qs; uint4* kl = (uint4*)ks; uint4* vl = (uint4*)vts;
#pragma unroll
    for (int i = 0; i < 2; ++i) {
      ql[tid + 128 * i] = qg[tid + 128 * i];
      kl[tid + 128 * i] = kg[tid + 128 * i];
      vl[tid + 128 * i] = vg[tid + 128 * i];
    }
  }
  if (tid < NTOK) {
    int ti = tid / 7, tj = tid - ti * 7;
    int hh = wi * 7 + ti, ww = wj * 7 + tj;
    int rh = hh < 49 ? 0 : (hh < 53 ? 1 : 2);
    int cw = ww < 49 ? 0 : (ww < 53 ? 1 : 2);
    lab[tid] = rh * 3 + cw;
  }
  __syncthreads();

  int wave = tid >> 5, lane = tid & 31, g = lane >> 4, nl = lane & 15;

  // S = q k^T   (B fragment of k^T == contiguous read of token-major k)
  {
    v16h a = load_frag_a(qs + wave * 16 * HDIM, HDIM, lane);
    v16h bb0 = load_frag_b(ks, HDIM, lane);
    v16h bb1 = load_frag_b(ks + 16 * HDIM, HDIM, lane);
    v16h bb2 = load_frag_b(ks + 32 * HDIM, HDIM, lane);
    v16h bb3 = load_frag_b(ks + 48 * HDIM, HDIM, lane);
    v8f s0 = {}, s1 = {}, s2 = {}, s3 = {};
    s0 = wmma16(a, bb0, s0);
    s1 = wmma16(a, bb1, s1);
    s2 = wmma16(a, bb2, s2);
    s3 = wmma16(a, bb3, s3);
#pragma unroll
    for (int r = 0; r < 8; ++r) {
      int row = (wave * 16 + g * 8 + r) * NTOKP;
      S[row + 0 * 16 + nl] = s0[r];
      S[row + 1 * 16 + nl] = s1[r];
      S[row + 2 * 16 + nl] = s2[r];
      S[row + 3 * 16 + nl] = s3[r];
    }
  }
  __syncthreads();

  // masked softmax, one thread per row
  if (tid < NTOKP) {
    int m = tid;
    if (m < NTOK) {
      int lm = lab[m];
      float mx = -1e30f;
      for (int n = 0; n < NTOK; ++n) {
        float s = S[m * NTOKP + n] + (lab[n] != lm ? -100.0f : 0.0f);
        S[m * NTOKP + n] = s;
        mx = fmaxf(mx, s);
      }
      float sum = 0.f;
      for (int n = 0; n < NTOK; ++n) sum += __expf(S[m * NTOKP + n] - mx);
      float inv = 1.0f / sum;
      for (int n = 0; n < NTOK; ++n)
        P[m * NTOKP + n] = (_Float16)(__expf(S[m * NTOKP + n] - mx) * inv);
      for (int n = NTOK; n < NTOKP; ++n) P[m * NTOKP + n] = (_Float16)0.0f;
    } else {
      for (int n = 0; n < NTOKP; ++n) P[m * NTOKP + n] = (_Float16)0.0f;
    }
  }
  __syncthreads();

  // O = P V  (B fragment of V == contiguous read of hd-major v^T)
  v8f oc[2] = {};
#pragma unroll
  for (int k0 = 0; k0 < NTOKP; k0 += 32) {
    v16h a = load_frag_a(P + wave * 16 * NTOKP + k0, NTOKP, lane);
#pragma unroll
    for (int t = 0; t < 2; ++t) {
      v16h b = load_frag_b(vts + t * 16 * NTOKP + k0, NTOKP, lane);
      oc[t] = wmma16(a, b, oc[t]);
    }
  }
#pragma unroll
  for (int t = 0; t < 2; ++t)
#pragma unroll
    for (int r = 0; r < 8; ++r) {
      int tok = wave * 16 + g * 8 + r;
      int hd = t * 16 + nl;
      o[((size_t)win * NTOKP + tok) * CDIM + head * HDIM + hd] = (_Float16)oc[t][r];
    }
}

// ---------------------------------------------------------------------------
// 4) proj GEMM + bias + window-reverse + residual -> x1 (f32)
// ---------------------------------------------------------------------------
__global__ __launch_bounds__(128) void gemm_proj_kernel(
    const _Float16* __restrict__ o, const _Float16* __restrict__ Wt,
    const float* __restrict__ bias, const float* __restrict__ x,
    float* __restrict__ x1) {
  int n0 = blockIdx.y * 64;
  int wave = threadIdx.x >> 5, lane = threadIdx.x & 31;
  const _Float16* A = o + (size_t)blockIdx.x * 128 * CDIM;
  v8f acc[8] = {};
  gemm_core_2x4(A, Wt + (size_t)n0 * CDIM, CDIM, CDIM, wave, lane, acc);
  int g = lane >> 4, nl = lane & 15;
#pragma unroll
  for (int p = 0; p < 8; ++p) {
    int c = n0 + 16 * (p & 3) + nl;
    float bb = bias[c];
#pragma unroll
    for (int r = 0; r < 8; ++r) {
      int mg = blockIdx.x * 128 + wave * 32 + (p >> 2) * 16 + g * 8 + r;
      int win = mg >> 6, tok = mg & 63;
      if (tok < NTOK) {
        int img = win >> 6, w_in = win & 63;
        int wi = w_in >> 3, wj = w_in & 7;
        int ti = tok / 7, tj = tok - ti * 7;
        int nsp = (wi * 7 + ti) * 56 + (wj * 7 + tj);
        size_t idx = ((size_t)img * 3136 + nsp) * CDIM + c;
        x1[idx] = x[idx] + acc[p][r] + bb;
      }
    }
  }
}

// ---------------------------------------------------------------------------
// 5) LN2 -> f16 [50176][384]
// ---------------------------------------------------------------------------
__global__ __launch_bounds__(256) void ln2_kernel(
    const float* __restrict__ x1, const float* __restrict__ gw,
    const float* __restrict__ bw, _Float16* __restrict__ h2) {
  int tok = blockIdx.x * 8 + (threadIdx.x >> 5);
  int lane = threadIdx.x & 31;
  const float* xr = x1 + (size_t)tok * CDIM;
  float vals[12], s = 0.f, s2 = 0.f;
#pragma unroll
  for (int i = 0; i < 12; ++i) {
    float v = xr[lane + 32 * i];
    vals[i] = v; s += v; s2 += v * v;
  }
  s = wave_sum(s); s2 = wave_sum(s2);
  float mu = s * (1.0f / CDIM);
  float rs = rsqrtf(s2 * (1.0f / CDIM) - mu * mu + 1e-5f);
  _Float16* hr = h2 + (size_t)tok * CDIM;
#pragma unroll
  for (int i = 0; i < 12; ++i) {
    int c = lane + 32 * i;
    hr[c] = (_Float16)((vals[i] - mu) * rs * gw[c] + bw[c]);
  }
}

// ---------------------------------------------------------------------------
// 6) fc1: [50176][384] x Wt[1536][384] + bias, GELU -> f16
// ---------------------------------------------------------------------------
__global__ __launch_bounds__(128) void gemm_fc1_kernel(
    const _Float16* __restrict__ h2, const _Float16* __restrict__ Wt,
    const float* __restrict__ bias, _Float16* __restrict__ out) {
  int m0 = blockIdx.x * 128, n0 = blockIdx.y * 64;
  int wave = threadIdx.x >> 5, lane = threadIdx.x & 31;
  const _Float16* A = h2 + (size_t)m0 * CDIM;
  v8f acc[8] = {};
  gemm_core_2x4(A, Wt + (size_t)n0 * CDIM, CDIM, CDIM, wave, lane, acc);
  int g = lane >> 4, nl = lane & 15;
#pragma unroll
  for (int p = 0; p < 8; ++p) {
    int c = n0 + 16 * (p & 3) + nl;
    float bb = bias[c];
#pragma unroll
    for (int r = 0; r < 8; ++r) {
      int m = m0 + wave * 32 + (p >> 2) * 16 + g * 8 + r;
      out[(size_t)m * 1536 + c] = (_Float16)gelu(acc[p][r] + bb);
    }
  }
}

// ---------------------------------------------------------------------------
// 7) fc2: [50176][1536] x Wt[384][1536] + bias, GELU, + x1 -> f32 out
// ---------------------------------------------------------------------------
__global__ __launch_bounds__(128) void gemm_fc2_kernel(
    const _Float16* __restrict__ a1, const _Float16* __restrict__ Wt,
    const float* __restrict__ bias, const float* __restrict__ x1,
    float* __restrict__ out) {
  int m0 = blockIdx.x * 128, n0 = blockIdx.y * 64;
  int wave = threadIdx.x >> 5, lane = threadIdx.x & 31;
  const _Float16* A = a1 + (size_t)m0 * 1536;
  v8f acc[8] = {};
  gemm_core_2x4(A, Wt + (size_t)n0 * 1536, 1536, 1536, wave, lane, acc);
  int g = lane >> 4, nl = lane & 15;
#pragma unroll
  for (int p = 0; p < 8; ++p) {
    int c = n0 + 16 * (p & 3) + nl;
    float bb = bias[c];
#pragma unroll
    for (int r = 0; r < 8; ++r) {
      int m = m0 + wave * 32 + (p >> 2) * 16 + g * 8 + r;
      size_t idx = (size_t)m * CDIM + c;
      out[idx] = x1[idx] + gelu(acc[p][r] + bb);
    }
  }
}

// ---------------------------------------------------------------------------
extern "C" void kernel_launch(void* const* d_in, const int* in_sizes, int n_in,
                              void* d_out, int out_size, void* d_ws, size_t ws_size,
                              hipStream_t stream) {
  (void)in_sizes; (void)n_in; (void)out_size; (void)ws_size;
  const float* x      = (const float*)d_in[0];
  const float* ln1_g  = (const float*)d_in[1];
  const float* ln1_b  = (const float*)d_in[2];
  const float* qkv_w  = (const float*)d_in[3];
  const float* proj_w = (const float*)d_in[4];
  const float* proj_b = (const float*)d_in[5];
  const float* ln2_g  = (const float*)d_in[6];
  const float* ln2_b  = (const float*)d_in[7];
  const float* fc1_w  = (const float*)d_in[8];
  const float* fc1_b  = (const float*)d_in[9];
  const float* fc2_w  = (const float*)d_in[10];
  const float* fc2_b  = (const float*)d_in[11];
  float* out = (float*)d_out;
  char* ws = (char*)d_ws;

  _Float16* qkvT = (_Float16*)(ws + OFF_QKVT);
  _Float16* projT = (_Float16*)(ws + OFF_PROJT);
  _Float16* fc1T = (_Float16*)(ws + OFF_FC1T);
  _Float16* fc2T = (_Float16*)(ws + OFF_FC2T);
  _Float16* yws  = (_Float16*)(ws + OFF_Y);
  _Float16* ows  = (_Float16*)(ws + OFF_Y);    // alias: y dead after QKV
  _Float16* qws  = (_Float16*)(ws + OFF_Q);
  _Float16* kws  = (_Float16*)(ws + OFF_K);
  _Float16* vtws = (_Float16*)(ws + OFF_VT);
  _Float16* h2ws = (_Float16*)(ws + OFF_Q);    // alias: q/k/v dead after attn
  float*    x1ws = (float*)(ws + OFF_X1);
  _Float16* a1ws = (_Float16*)(ws + OFF_FC1O);

  wt_cvt_kernel<<<(384 * 1152 + 255) / 256, 256, 0, stream>>>(qkv_w, qkvT, 384, 1152);
  wt_cvt_kernel<<<(384 * 384 + 255) / 256, 256, 0, stream>>>(proj_w, projT, 384, 384);
  wt_cvt_kernel<<<(384 * 1536 + 255) / 256, 256, 0, stream>>>(fc1_w, fc1T, 384, 1536);
  wt_cvt_kernel<<<(1536 * 384 + 255) / 256, 256, 0, stream>>>(fc2_w, fc2T, 1536, 384);

  ln1_window_kernel<<<NWINB, 256, 0, stream>>>(x, ln1_g, ln1_b, yws);
  gemm_qkv_kernel<<<dim3(NWINB * NTOKP / 128, 18), 128, 0, stream>>>(yws, qkvT, qws, kws, vtws);
  attn_kernel<<<NWINB * HEADS, 128, 0, stream>>>(qws, kws, vtws, ows);
  gemm_proj_kernel<<<dim3(NWINB * NTOKP / 128, 6), 128, 0, stream>>>(ows, projT, proj_b, x, x1ws);
  ln2_kernel<<<NTOKENS / 8, 256, 0, stream>>>(x1ws, ln2_g, ln2_b, h2ws);
  gemm_fc1_kernel<<<dim3(NTOKENS / 128, 24), 128, 0, stream>>>(h2ws, fc1T, fc1_b, a1ws);
  gemm_fc2_kernel<<<dim3(NTOKENS / 128, 6), 128, 0, stream>>>(a1ws, fc2T, fc2_b, x1ws, out);
}
